// MultiHeadedAttention_14233521619291
// MI455X (gfx1250) — matrix operand order repaired and checked by reading, not compiled, not run on hardware
//
#include <hip/hip_runtime.h>
#include <hip/hip_bf16.h>

// ---------------- constants (match reference) ----------------
#define BATCH 4
#define SEQ   2048
#define DMODEL 1024
#define NHEAD 16
#define DKH   64

typedef _Float16 v16h __attribute__((ext_vector_type(16)));
typedef float    v8f  __attribute__((ext_vector_type(8)));
typedef int      b128v __attribute__((vector_size(4 * sizeof(int))));

// ---------------- async global->LDS (gfx1250) ----------------
#if __has_builtin(__builtin_amdgcn_global_load_async_to_lds_b128) && \
    __has_builtin(__builtin_amdgcn_s_wait_asynccnt)
#define ASYNC_LDS 1
#else
#define ASYNC_LDS 0
#endif

static __device__ __forceinline__ void cp128(void* dst_lds, const void* src_g) {
#if ASYNC_LDS
  __builtin_amdgcn_global_load_async_to_lds_b128(
      (__attribute__((address_space(1))) b128v*)(__UINTPTR_TYPE__)src_g,
      (__attribute__((address_space(3))) b128v*)(unsigned int)(__UINTPTR_TYPE__)dst_lds,
      0, 0);
#else
  *(uint4*)dst_lds = *(const uint4*)src_g;
#endif
}

static __device__ __forceinline__ void cp_wait() {
#if ASYNC_LDS
  __builtin_amdgcn_s_wait_asynccnt(0);
#endif
}

static __device__ __forceinline__ v8f zero_v8f() {
  v8f z;
#pragma unroll
  for (int i = 0; i < 8; ++i) z[i] = 0.0f;
  return z;
}

static __device__ __forceinline__ v8f wmma_f16(v16h a, v16h b, v8f c) {
  return __builtin_amdgcn_wmma_f32_16x16x32_f16(
      false, a, false, b, (short)0, c, false, false);
}

// ---- fragment loaders (wave32), contiguous k = 16*(lane/16) + i in A and B ----
static __device__ __forceinline__ v16h load_a(const _Float16* src, int ld) {
  const int lane = threadIdx.x & 31;
  const int m = lane & 15;
  const int grp = lane >> 4;
  const _Float16* p = src + m * ld + grp * 16;
  v16h a;
#pragma unroll
  for (int i = 0; i < 16; ++i) a[i] = p[i];
  return a;
}

static __device__ __forceinline__ v16h load_b_nt(const _Float16* src, int ld) {
  const int lane = threadIdx.x & 31;
  const int n = lane & 15;
  const int grp = lane >> 4;
  const _Float16* p = src + n * ld + grp * 16;
  v16h b;
#pragma unroll
  for (int i = 0; i < 16; ++i) b[i] = p[i];
  return b;
}

static __device__ __forceinline__ v16h load_b_nn(const _Float16* src, int ld) {
  const int lane = threadIdx.x & 31;
  const int n = lane & 15;
  const int grp = lane >> 4;
  const _Float16* p = src + (grp * 16) * ld + n;
  v16h b;
#pragma unroll
  for (int i = 0; i < 16; ++i) b[i] = p[i * ld];
  return b;
}

// ================= f32 -> f16 conversion (one pass) =================
__global__ __launch_bounds__(256) void cvt_f16(const float* __restrict__ src,
                                               _Float16* __restrict__ dst, int n4) {
  int i = blockIdx.x * blockDim.x + threadIdx.x;
  if (i < n4) {
    const float4 v = *(const float4*)(src + (size_t)i * 4);
    union { _Float16 h[4]; uint2 u; } o;
    o.h[0] = (_Float16)v.x; o.h[1] = (_Float16)v.y;
    o.h[2] = (_Float16)v.z; o.h[3] = (_Float16)v.w;
    *(uint2*)(dst + (size_t)i * 4) = o.u;
  }
}

// ================= GEMM: Y = X * W^T + bias (X, W pre-converted f16) ========
// OUT_MODE 0: f16 out in [B,H,S,DK] layout, value = (acc+bias)*scale
// OUT_MODE 1: f32 out flat [8192 x 1024], value = acc+bias
#define GBM 128
#define GBN 64
#define GBK 32
#define GLD 40  // LDS stride (halves); 40*2=80B, 16B aligned rows

template <int OUT_MODE>
__global__ __launch_bounds__(128) void gemm_nt(const _Float16* __restrict__ X,
                                               const _Float16* __restrict__ W,
                                               const float* __restrict__ bias,
                                               void* __restrict__ out_, float scale) {
  __shared__ alignas(16) _Float16 xs[2][GBM * GLD];
  __shared__ alignas(16) _Float16 ws[2][GBN * GLD];
  const int tid = threadIdx.x;
  const int wave = tid >> 5;
  const int lane = tid & 31;
  const int n = lane & 15;
  const int grp = lane >> 4;
  const int row0 = blockIdx.x * GBM;
  const int col0 = blockIdx.y * GBN;
  const int wm = wave * 32;  // 32 rows per wave, all 64 cols

  auto stage = [&](int buf, int k0) {
    // X tile: 128 rows x 32 halves = 512 x 16B chunks
#pragma unroll
    for (int e = tid; e < GBM * 4; e += 128) {
      const int r = e >> 2, c = (e & 3) * 8;
      cp128(&xs[buf][r * GLD + c], &X[(size_t)(row0 + r) * DMODEL + k0 + c]);
    }
    // W tile: 64 rows x 32 halves = 256 chunks
#pragma unroll
    for (int e = tid; e < GBN * 4; e += 128) {
      const int r = e >> 2, c = (e & 3) * 8;
      cp128(&ws[buf][r * GLD + c], &W[(size_t)(col0 + r) * DMODEL + k0 + c]);
    }
  };

  v8f acc[2][4];
#pragma unroll
  for (int i = 0; i < 2; ++i)
#pragma unroll
    for (int j = 0; j < 4; ++j) acc[i][j] = zero_v8f();

  stage(0, 0);
  for (int kb = 0; kb < DMODEL / GBK; ++kb) {
    const int cur = kb & 1;
    cp_wait();
    __syncthreads();  // buffer `cur` visible to all waves
    if (kb + 1 < DMODEL / GBK) stage(cur ^ 1, (kb + 1) * GBK);

    const v16h a0 = load_a(xs[cur] + (size_t)wm * GLD, GLD);
    const v16h a1 = load_a(xs[cur] + (size_t)(wm + 16) * GLD, GLD);
#pragma unroll
    for (int j = 0; j < 4; ++j) {
      const v16h b = load_b_nt(ws[cur] + (size_t)(j * 16) * GLD, GLD);
      acc[0][j] = wmma_f16(a0, b, acc[0][j]);
      acc[1][j] = wmma_f16(a1, b, acc[1][j]);
    }
    __syncthreads();  // all waves done reading `cur` before it is restaged
  }

#pragma unroll
  for (int i = 0; i < 2; ++i) {
#pragma unroll
    for (int j = 0; j < 4; ++j) {
      const int col = col0 + j * 16 + n;
      const float bv = bias[col];
#pragma unroll
      for (int r = 0; r < 8; ++r) {
        const int row = row0 + wm + i * 16 + grp * 8 + r;
        const float v = acc[i][j][r] + bv;
        if (OUT_MODE == 0) {
          const int bb = row >> 11;            // /SEQ
          const int s  = row & (SEQ - 1);
          const int hh = col >> 6;             // /DKH
          const int dk = col & (DKH - 1);
          ((_Float16*)out_)[((((size_t)bb * NHEAD + hh) * SEQ + s) << 6) + dk] =
              (_Float16)(v * scale);
        } else {
          ((float*)out_)[(size_t)row * DMODEL + col] = v;
        }
      }
    }
  }
}

// ================= Flash attention over f16 Q/K/V [B,H,S,DK] =================
#define AQ 64    // query rows per block
#define AK 32    // keys per block-step
#define LQK 72   // LDS stride (halves); 144B rows, 16B aligned
#define LSC 33   // score tile stride (floats)
#define LP  40   // P tile stride (halves)

__global__ __launch_bounds__(128) void attn_kernel(const _Float16* __restrict__ Q,
                                                   const _Float16* __restrict__ K,
                                                   const _Float16* __restrict__ V,
                                                   const int* __restrict__ mask,
                                                   _Float16* __restrict__ ctx) {
  __shared__ alignas(16) _Float16 qs[AQ * LQK];
  __shared__ alignas(16) _Float16 ks[2][AK * LQK];
  __shared__ alignas(16) _Float16 vs[2][AK * LQK];
  __shared__ float sc[AQ * LSC];
  __shared__ alignas(16) _Float16 ps[AQ * LP];
  __shared__ float a_s[AQ];
  __shared__ float l_s[AQ];

  const int tid = threadIdx.x;
  const int wave = tid >> 5;
  const int lane = tid & 31;
  const int n = lane & 15;
  const int grp = lane >> 4;
  const int q0 = blockIdx.x * AQ;
  const int h = blockIdx.y;
  const int b = blockIdx.z;
  const int m0 = wave * 16;

  const size_t head_off = (((size_t)b * NHEAD + h) * SEQ) * DKH;
  const _Float16* Qh = Q + head_off;
  const _Float16* Kh = K + head_off;
  const _Float16* Vh = V + head_off;

  auto stageKV = [&](int buf, int kb0) {
    // 32 rows x 64 halves = 256 chunks per matrix
#pragma unroll
    for (int e = tid; e < AK * 8; e += 128) {
      const int r = e >> 3, c = (e & 7) * 8;
      cp128(&ks[buf][r * LQK + c], &Kh[(size_t)(kb0 + r) * DKH + c]);
      cp128(&vs[buf][r * LQK + c], &Vh[(size_t)(kb0 + r) * DKH + c]);
    }
  };

  // stage Q tile (64 x 64 halves = 512 chunks) + first K/V block
#pragma unroll
  for (int e = tid; e < AQ * 8; e += 128) {
    const int r = e >> 3, c = (e & 7) * 8;
    cp128(&qs[r * LQK + c], &Qh[(size_t)(q0 + r) * DKH + c]);
  }
  stageKV(0, 0);

  v8f o[4];
#pragma unroll
  for (int t = 0; t < 4; ++t) o[t] = zero_v8f();

  float m_run = -INFINITY;  // meaningful for tid<64
  float l_run = 0.0f;

  for (int kb = 0; kb < SEQ / AK; ++kb) {
    const int cur = kb & 1;
    cp_wait();
    __syncthreads();  // qs + ks/vs[cur] ready everywhere
    if (kb + 1 < SEQ / AK) stageKV(cur ^ 1, (kb + 1) * AK);

    // S = Q * K^T  (per wave: 16 q-rows x 32 keys)
    {
      const v16h a0 = load_a(qs + (size_t)m0 * LQK, LQK);
      const v16h a1 = load_a(qs + (size_t)m0 * LQK + 32, LQK);
#pragma unroll
      for (int j = 0; j < 2; ++j) {
        v8f s = zero_v8f();
        const v16h bk0 = load_b_nt(ks[cur] + (size_t)(j * 16) * LQK, LQK);
        const v16h bk1 = load_b_nt(ks[cur] + (size_t)(j * 16) * LQK + 32, LQK);
        s = wmma_f16(a0, bk0, s);
        s = wmma_f16(a1, bk1, s);
#pragma unroll
        for (int r = 0; r < 8; ++r)
          sc[(m0 + grp * 8 + r) * LSC + j * 16 + n] = s[r];
      }
    }
    __syncthreads();

    // online softmax: one thread per query row
    if (tid < 64) {
      const int row = tid;
      const int* mrow = mask + ((size_t)b * SEQ + (q0 + row)) * SEQ + kb * AK;
      float sv[AK];
      float rmax = -INFINITY;
#pragma unroll
      for (int j = 0; j < AK; ++j) {
        float x = sc[row * LSC + j];
        if (mrow[j] == 0) x = -1e9f;
        sv[j] = x;
        rmax = fmaxf(rmax, x);
      }
      const float mnew = fmaxf(m_run, rmax);
      const float alpha = __expf(m_run - mnew);  // exp(-inf)=0 on first block
      float ssum = 0.0f;
#pragma unroll
      for (int j = 0; j < AK; ++j) {
        float p = __expf(sv[j] - mnew);
        ps[row * LP + j] = (_Float16)p;
        ssum += p;
      }
      l_run = l_run * alpha + ssum;
      m_run = mnew;
      a_s[row] = alpha;
      l_s[row] = l_run;
    }
    __syncthreads();

    // O = O*alpha + P * V
#pragma unroll
    for (int t = 0; t < 4; ++t)
#pragma unroll
      for (int r = 0; r < 8; ++r) o[t][r] *= a_s[m0 + grp * 8 + r];

    {
      const v16h pa = load_a(ps + (size_t)m0 * LP, LP);
#pragma unroll
      for (int t = 0; t < 4; ++t) {
        const v16h bv = load_b_nn(vs[cur] + t * 16, LQK);
        o[t] = wmma_f16(pa, bv, o[t]);
      }
    }
    __syncthreads();  // consumers done with buffer `cur` before restage
  }

  // normalize, write ctx [B,S,D] f16
#pragma unroll
  for (int t = 0; t < 4; ++t) {
#pragma unroll
    for (int r = 0; r < 8; ++r) {
      const int row = m0 + grp * 8 + r;
      const float l = l_s[row];
      const float inv = l > 0.0f ? 1.0f / l : 0.0f;
      ctx[((size_t)b * SEQ + q0 + row) * DMODEL + h * DKH + t * 16 + n] =
          (_Float16)(o[t][r] * inv);
    }
  }
}

// ================= host launcher =================
extern "C" void kernel_launch(void* const* d_in, const int* in_sizes, int n_in,
                              void* d_out, int out_size, void* d_ws, size_t ws_size,
                              hipStream_t stream) {
  const float* query = (const float*)d_in[0];
  const float* key   = (const float*)d_in[1];
  const float* value = (const float*)d_in[2];
  const int*   mask  = (const int*)d_in[3];
  const float* Wq = (const float*)d_in[4];
  const float* bq = (const float*)d_in[5];
  const float* Wk = (const float*)d_in[6];
  const float* bk = (const float*)d_in[7];
  const float* Wv = (const float*)d_in[8];
  const float* bv = (const float*)d_in[9];
  const float* Wo = (const float*)d_in[10];
  const float* bo = (const float*)d_in[11];

  const size_t xel = (size_t)BATCH * SEQ * DMODEL;  // 8.39M
  const size_t wel = (size_t)DMODEL * DMODEL;       // 1.05M
  _Float16* p = (_Float16*)d_ws;
  _Float16* xq_h = p; p += xel;
  _Float16* xk_h = p; p += xel;
  _Float16* xv_h = p; p += xel;
  _Float16* wq_h = p; p += wel;
  _Float16* wk_h = p; p += wel;
  _Float16* wv_h = p; p += wel;
  _Float16* wo_h = p; p += wel;
  _Float16* qbuf = p; p += xel;
  _Float16* kbuf = p; p += xel;
  _Float16* vbuf = p; p += xel;
  _Float16* ctx  = p; p += xel;

  // one-pass f32 -> f16 conversion (inputs + weights)
  {
    const int nx4 = (int)(xel / 4), nw4 = (int)(wel / 4);
    cvt_f16<<<(nx4 + 255) / 256, 256, 0, stream>>>(query, xq_h, nx4);
    cvt_f16<<<(nx4 + 255) / 256, 256, 0, stream>>>(key,   xk_h, nx4);
    cvt_f16<<<(nx4 + 255) / 256, 256, 0, stream>>>(value, xv_h, nx4);
    cvt_f16<<<(nw4 + 255) / 256, 256, 0, stream>>>(Wq, wq_h, nw4);
    cvt_f16<<<(nw4 + 255) / 256, 256, 0, stream>>>(Wk, wk_h, nw4);
    cvt_f16<<<(nw4 + 255) / 256, 256, 0, stream>>>(Wv, wv_h, nw4);
    cvt_f16<<<(nw4 + 255) / 256, 256, 0, stream>>>(Wo, wo_h, nw4);
  }

  const dim3 gproj((BATCH * SEQ) / GBM, DMODEL / GBN);  // 64 x 16
  const float qscale = 0.125f;  // 1/sqrt(DK), folded into Q

  gemm_nt<0><<<gproj, 128, 0, stream>>>(xq_h, wq_h, bq, qbuf, qscale);
  gemm_nt<0><<<gproj, 128, 0, stream>>>(xk_h, wk_h, bk, kbuf, 1.0f);
  gemm_nt<0><<<gproj, 128, 0, stream>>>(xv_h, wv_h, bv, vbuf, 1.0f);

  const dim3 gattn(SEQ / AQ, NHEAD, BATCH);  // 32 x 16 x 4
  attn_kernel<<<gattn, 128, 0, stream>>>(qbuf, kbuf, vbuf, mask, ctx);

  gemm_nt<1><<<gproj, 128, 0, stream>>>(ctx, wo_h, bo, d_out, 1.0f);
}
